// GPSFlexBlock_88493506166792
// MI455X (gfx1250) — compile-verified, matrix operand order checked by reading, not run on hardware
//
#include <hip/hip_runtime.h>
#include <hip/hip_bf16.h>
#include <cstddef>

// GPS/GRIT block for MI455X (gfx1250, wave32).
// All dense GEMMs run on v_wmma_f32_16x16x32_bf16 (f32 accumulate).
// N=50000 and E=600000 are multiples of 16 -> exact 16x16 tiling.

#define WDIM   128
#define WATTN  64
#define WHID   256
#define NGRAPHS 64

typedef __attribute__((ext_vector_type(16))) __bf16 v16bf;
typedef __attribute__((ext_vector_type(8)))  float  v8f;

// ---------------------------------------------------------------- utilities

__device__ __forceinline__ float wave_sum(float v) {
#pragma unroll
  for (int off = 16; off > 0; off >>= 1) v += __shfl_xor(v, off, 32);
  return v;
}

// One 16x16 output tile: acc += A[rowBase:rowBase+16, 0:Kin] * Bt[colBase:colBase+16, 0:Kin]^T
// A is MxKin row-major bf16; Bt is KoutxKin row-major bf16 (i.e. W transposed),
// so both operand fragments are contiguous 2xbf16 dword loads along K.
// Fragment layout per CDNA5 ISA 7.12.2 (16-bit A 16x32): lane = idx%16,
// hi-half lanes carry K offset +8; VGPR i covers K = (i/4)*16 + hi*8 + (i%4)*2.
__device__ __forceinline__ v8f wmma_bf16_tiles(const __bf16* __restrict__ A, int lda, int rowBase,
                                               const __bf16* __restrict__ Bt, int ldb, int colBase,
                                               int Kin, v8f acc) {
  const int l  = threadIdx.x & 31;
  const int r  = l & 15;
  const int hi = l >> 4;
  const unsigned* __restrict__ Arow = (const unsigned*)(A + (size_t)(rowBase + r) * lda);
  const unsigned* __restrict__ Bcol = (const unsigned*)(Bt + (size_t)(colBase + r) * ldb);
  const int kPairs = Kin >> 1;            // pairs of bf16 per row
  for (int kb = 0; kb < kPairs; kb += 16) {   // 16 pairs = 32 K per WMMA
    union { v16bf v; unsigned u[8]; } a, b;
#pragma unroll
    for (int i = 0; i < 8; ++i) {
      int k2 = kb + ((i >> 2) << 3) + (hi << 2) + (i & 3);
      a.u[i] = Arow[k2];
      b.u[i] = Bcol[k2];
    }
    __builtin_prefetch(Arow + kb + 16, 0, 1);   // global_prefetch_b8
    acc = __builtin_amdgcn_wmma_f32_16x16x32_bf16(
        /*neg_a=*/false, a.v, /*neg_b=*/false, b.v,
        /*c_mod=*/(short)0, acc, /*reuse_a=*/false, /*reuse_b=*/false);
  }
  return acc;
}

// ---------------------------------------------------------------- kernels

// fp32 W (Kin x Kout) -> bf16 W^T (Kout x Kin)
__global__ void convert_wT_kernel(const float* __restrict__ W, __bf16* __restrict__ Wt,
                                  int Kin, int Kout) {
  int i = blockIdx.x * blockDim.x + threadIdx.x;
  if (i >= Kin * Kout) return;
  int k = i / Kout, j = i % Kout;
  Wt[(size_t)j * Kin + k] = (__bf16)W[i];
}

// per-graph sums / counts (wave per node row, float4 per lane)
__global__ void seg_sum_kernel(const float* __restrict__ X, const int* __restrict__ bidx,
                               float* __restrict__ gsum, float* __restrict__ gcnt, int N) {
  int row  = blockIdx.x * 8 + (threadIdx.x >> 5);
  int lane = threadIdx.x & 31;
  if (row >= N) return;
  int g = bidx[row];
  float4 v = ((const float4*)(X + (size_t)row * WDIM))[lane];
  float* p = gsum + (size_t)g * WDIM + lane * 4;
  atomicAdd(p + 0, v.x); atomicAdd(p + 1, v.y);
  atomicAdd(p + 2, v.z); atomicAdd(p + 3, v.w);
  if (lane == 0) atomicAdd(&gcnt[g], 1.0f);
}

// g2[b,:] = (g_token + gsum[b]/max(cnt,1)) @ Wg + Wg_b   (tiny 64x128x128)
__global__ void graph_linear_kernel(const float* __restrict__ gsum, const float* __restrict__ gcnt,
                                    const float* __restrict__ g_token, const float* __restrict__ Wg,
                                    const float* __restrict__ Wgb, float* __restrict__ g2) {
  int b = blockIdx.x, j = threadIdx.x;
  float cnt = gcnt[b]; cnt = cnt > 1.0f ? cnt : 1.0f;
  float acc = Wgb[j];
  for (int k = 0; k < WDIM; ++k)
    acc += (g_token[k] + gsum[(size_t)b * WDIM + k] / cnt) * Wg[(size_t)k * WDIM + j];
  g2[(size_t)b * WDIM + j] = acc;
}

// Xcur = X + g2[batch_idx]
__global__ void add_gather_kernel(const float* __restrict__ X, const float* __restrict__ g2,
                                  const int* __restrict__ bidx, float* __restrict__ out, int N) {
  size_t i = (size_t)blockIdx.x * blockDim.x + threadIdx.x;
  if (i >= (size_t)N * WDIM) return;
  int row = (int)(i >> 7), col = (int)(i & 127);
  out[i] = X[i] + g2[(size_t)bidx[row] * WDIM + col];
}

// LayerNorm over 128 features; wave per row; writes optional f32 and bf16 outputs.
__global__ void layernorm_kernel(const float* __restrict__ Xin, const float* __restrict__ g,
                                 const float* __restrict__ b, float* __restrict__ outF,
                                 __bf16* __restrict__ outBf, int N) {
  int row  = blockIdx.x * 8 + (threadIdx.x >> 5);
  int lane = threadIdx.x & 31;
  if (row >= N) return;
  float4 v = ((const float4*)(Xin + (size_t)row * WDIM))[lane];
  float mean = wave_sum(v.x + v.y + v.z + v.w) * (1.0f / WDIM);
  float d0 = v.x - mean, d1 = v.y - mean, d2 = v.z - mean, d3 = v.w - mean;
  float var = wave_sum(d0 * d0 + d1 * d1 + d2 * d2 + d3 * d3) * (1.0f / WDIM);
  float rstd = rsqrtf(var + 1e-5f);
  int c = lane * 4;
  float y0 = d0 * rstd * g[c + 0] + b[c + 0];
  float y1 = d1 * rstd * g[c + 1] + b[c + 1];
  float y2 = d2 * rstd * g[c + 2] + b[c + 2];
  float y3 = d3 * rstd * g[c + 3] + b[c + 3];
  size_t base = (size_t)row * WDIM + c;
  if (outF) { outF[base] = y0; outF[base + 1] = y1; outF[base + 2] = y2; outF[base + 3] = y3; }
  outBf[base] = (__bf16)y0; outBf[base + 1] = (__bf16)y1;
  outBf[base + 2] = (__bf16)y2; outBf[base + 3] = (__bf16)y3;
}

// Generic WMMA GEMM: out = [residual +] [relu](A @ W + bias); optional bf16 copy.
// A: MxKin bf16 row-major, Bt: KoutxKin bf16. Block = 8 waves, wave -> one 16x16 tile.
__global__ void wmma_gemm_kernel(const __bf16* __restrict__ A, const __bf16* __restrict__ Bt,
                                 const float* __restrict__ bias, const float* __restrict__ residual,
                                 float* __restrict__ outF, __bf16* __restrict__ outBf,
                                 int M, int Kin, int Kout, int relu) {
  int colTile = blockIdx.y * 8 + (threadIdx.x >> 5);
  if (colTile * 16 >= Kout) return;
  int rowBase = blockIdx.x * 16;
  int colBase = colTile * 16;
  v8f acc = {};
  acc = wmma_bf16_tiles(A, Kin, rowBase, Bt, Kin, colBase, Kin, acc);
  const int l = threadIdx.x & 31, r = l & 15, hi = l >> 4;
  const int col = colBase + r;
  const float bv = bias ? bias[col] : 0.0f;
#pragma unroll
  for (int i = 0; i < 8; ++i) {           // C/D layout: VGPR i -> row i (+8 for hi half)
    int row = rowBase + i + hi * 8;
    float v = acc[i] + bv;
    if (relu) v = v > 0.0f ? v : 0.0f;
    size_t idx = (size_t)row * Kout + col;
    if (residual) v += residual[idx];
    if (outF)  outF[idx]  = v;
    if (outBf) outBf[idx] = (__bf16)v;
  }
}

// agg[src[e]] += Mf[dst[e]]   (wave per edge, float4 per lane)
__global__ void scatter_agg_kernel(const float* __restrict__ Mf, const int* __restrict__ srcIdx,
                                   const int* __restrict__ dstIdx, float* __restrict__ agg, int E) {
  int e    = (int)(((size_t)blockIdx.x * blockDim.x + threadIdx.x) >> 5);
  int lane = threadIdx.x & 31;
  if (e >= E) return;
  int s = srcIdx[e], d = dstIdx[e];
  float4 v = ((const float4*)(Mf + (size_t)d * WDIM))[lane];
  float* p = agg + (size_t)s * WDIM + lane * 4;
  atomicAdd(p + 0, v.x); atomicAdd(p + 1, v.y);
  atomicAdd(p + 2, v.z); atomicAdd(p + 3, v.w);
}

// tmpBf = bf16((1+eps)*Xn + agg)
__global__ void gin_pre_kernel(const float* __restrict__ Xn, const float* __restrict__ agg,
                               const float* __restrict__ epsp, __bf16* __restrict__ outBf,
                               size_t total) {
  size_t i = (size_t)blockIdx.x * blockDim.x + threadIdx.x;
  if (i >= total) return;
  outBf[i] = (__bf16)((1.0f + epsp[0]) * Xn[i] + agg[i]);
}

// e = tanh(Q[src]+K[dst]) -> bf16; scores = e . Wphi   (wave per edge, 2 feats/lane)
__global__ void edge_e_kernel(const float* __restrict__ Qf, const float* __restrict__ Kf,
                              const float* __restrict__ Wphi, const int* __restrict__ srcIdx,
                              const int* __restrict__ dstIdx, __bf16* __restrict__ eBf,
                              float* __restrict__ scores, int E) {
  int e    = (int)(((size_t)blockIdx.x * blockDim.x + threadIdx.x) >> 5);
  int lane = threadIdx.x & 31;
  if (e >= E) return;
  int s = srcIdx[e], d = dstIdx[e];
  float2 q = ((const float2*)(Qf + (size_t)s * WATTN))[lane];
  float2 k = ((const float2*)(Kf + (size_t)d * WATTN))[lane];
  float2 w = ((const float2*)Wphi)[lane];
  float e0 = tanhf(q.x + k.x);
  float e1 = tanhf(q.y + k.y);
  size_t base = (size_t)e * WATTN + 2 * lane;
  eBf[base] = (__bf16)e0; eBf[base + 1] = (__bf16)e1;
  float sc = wave_sum(e0 * w.x + e1 * w.y);
  if (lane == 0) scores[e] = sc;
}

// denom[src[e]] += exp(scores[e])
__global__ void denom_kernel(const float* __restrict__ scores, const int* __restrict__ srcIdx,
                             float* __restrict__ denom, int E) {
  int e = blockIdx.x * blockDim.x + threadIdx.x;
  if (e >= E) return;
  atomicAdd(&denom[srcIdx[e]], __expf(scores[e]));
}

// WMMA over edge tiles: ev = e @ We^T; X[src] += alpha*(Y[dst] + ev + We_b)  (fused scatter)
__global__ void attn_apply_kernel(const __bf16* __restrict__ eBf, const __bf16* __restrict__ WeT,
                                  const float* __restrict__ Web, const float* __restrict__ Yf,
                                  const float* __restrict__ scores, const float* __restrict__ denom,
                                  const int* __restrict__ srcIdx, const int* __restrict__ dstIdx,
                                  float* __restrict__ Xout, int E) {
  int rowBase = blockIdx.x * 16;
  int colBase = (threadIdx.x >> 5) * 16;        // 8 waves x 16 = 128 cols
  v8f acc = {};
  acc = wmma_bf16_tiles(eBf, WATTN, rowBase, WeT, WATTN, colBase, WATTN, acc);
  const int l = threadIdx.x & 31, r = l & 15, hi = l >> 4;
  const int col = colBase + r;
  const float wb = Web[col];
#pragma unroll
  for (int i = 0; i < 8; ++i) {
    int e = rowBase + i + hi * 8;
    int s = srcIdx[e], d = dstIdx[e];
    float alpha = __expf(scores[e]) / (denom[s] + 1e-9f);
    float val = alpha * (Yf[(size_t)d * WDIM + col] + acc[i] + wb);
    atomicAdd(&Xout[(size_t)s * WDIM + col], val);
  }
}

// ---------------------------------------------------------------- host launcher

extern "C" void kernel_launch(void* const* d_in, const int* in_sizes, int n_in,
                              void* d_out, int out_size, void* d_ws, size_t ws_size,
                              hipStream_t stream) {
  (void)n_in; (void)out_size; (void)ws_size;
  const int N = in_sizes[0] / WDIM;       // 50000
  const int E = in_sizes[1] / 2;          // 600000
  const int B = NGRAPHS;                  // 64 (fixed by reference)

  const float* X       = (const float*)d_in[0];
  const int*   eidx    = (const int*)d_in[1];
  const int*   src     = eidx;            // edge_index[0]
  const int*   dst     = eidx + E;        // edge_index[1]
  const int*   bidx    = (const int*)d_in[2];
  const float* g_token = (const float*)d_in[4];
  const float* Wg_w    = (const float*)d_in[5];
  const float* Wg_b    = (const float*)d_in[6];
  const float* ln1_g   = (const float*)d_in[7];
  const float* ln1_b   = (const float*)d_in[8];
  const float* eps     = (const float*)d_in[9];
  const float* msg_w   = (const float*)d_in[10];
  const float* msg_b   = (const float*)d_in[11];
  const float* lin_w   = (const float*)d_in[12];
  const float* lin_b   = (const float*)d_in[13];
  const float* ln2_g   = (const float*)d_in[14];
  const float* ln2_b   = (const float*)d_in[15];
  const float* Wq      = (const float*)d_in[16];
  const float* Wk      = (const float*)d_in[17];
  const float* Wphi    = (const float*)d_in[18];
  const float* Wy_w    = (const float*)d_in[19];
  const float* Wy_b    = (const float*)d_in[20];
  const float* We_w    = (const float*)d_in[21];
  const float* We_b    = (const float*)d_in[22];
  const float* ln3_g   = (const float*)d_in[23];
  const float* ln3_b   = (const float*)d_in[24];
  const float* ffn_w1  = (const float*)d_in[25];
  const float* ffn_b1  = (const float*)d_in[26];
  const float* ffn_w2  = (const float*)d_in[27];
  const float* ffn_b2  = (const float*)d_in[28];

  float* Xcur = (float*)d_out;            // running residual stream (N x 128)

  // ---- carve workspace (256B-aligned regions)
  char* w = (char*)d_ws;
  auto carve = [&](size_t bytes) { char* p = w; w += (bytes + 255) & ~(size_t)255; return p; };
  float*  gsum  = (float*)carve((size_t)B * WDIM * 4);
  float*  gcnt  = (float*)carve((size_t)B * 4);
  float*  g2    = (float*)carve((size_t)B * WDIM * 4);
  float*  Xn    = (float*)carve((size_t)N * WDIM * 4);
  float*  MfYf  = (float*)carve((size_t)N * WDIM * 4);   // msg values, later Y
  float*  agg   = (float*)carve((size_t)N * WDIM * 4);
  float*  Qf    = (float*)carve((size_t)N * WATTN * 4);
  float*  Kf    = (float*)carve((size_t)N * WATTN * 4);
  float*  scors = (float*)carve((size_t)E * 4);
  float*  denom = (float*)carve((size_t)N * 4);
  __bf16* XnBf  = (__bf16*)carve((size_t)N * WDIM * 2);
  __bf16* HBf   = (__bf16*)carve((size_t)N * WHID * 2);
  __bf16* eBf   = (__bf16*)carve((size_t)E * WATTN * 2);
  __bf16* msgT  = (__bf16*)carve((size_t)WDIM * WDIM * 2);
  __bf16* linT  = (__bf16*)carve((size_t)WDIM * WDIM * 2);
  __bf16* WqT   = (__bf16*)carve((size_t)WDIM * WATTN * 2);
  __bf16* WkT   = (__bf16*)carve((size_t)WDIM * WATTN * 2);
  __bf16* WyT   = (__bf16*)carve((size_t)WDIM * WDIM * 2);
  __bf16* WeT   = (__bf16*)carve((size_t)WATTN * WDIM * 2);
  __bf16* W1T   = (__bf16*)carve((size_t)WDIM * WHID * 2);
  __bf16* W2T   = (__bf16*)carve((size_t)WHID * WDIM * 2);

  // ---- zero the accumulators (poisoned between runs)
  hipMemsetAsync(gsum,  0, (size_t)B * WDIM * 4, stream);
  hipMemsetAsync(gcnt,  0, (size_t)B * 4, stream);
  hipMemsetAsync(agg,   0, (size_t)N * WDIM * 4, stream);
  hipMemsetAsync(denom, 0, (size_t)N * 4, stream);

  // ---- weight conversions (fp32 -> transposed bf16)
  auto convW = [&](const float* Wp, __bf16* Wt, int Kin, int Kout) {
    int tot = Kin * Kout;
    convert_wT_kernel<<<(tot + 255) / 256, 256, 0, stream>>>(Wp, Wt, Kin, Kout);
  };
  convW(msg_w,  msgT, WDIM, WDIM);
  convW(lin_w,  linT, WDIM, WDIM);
  convW(Wq,     WqT,  WDIM, WATTN);
  convW(Wk,     WkT,  WDIM, WATTN);
  convW(Wy_w,   WyT,  WDIM, WDIM);
  convW(We_w,   WeT,  WATTN, WDIM);
  convW(ffn_w1, W1T,  WDIM, WHID);
  convW(ffn_w2, W2T,  WHID, WDIM);

  const int rowBlocks  = (N + 7) / 8;      // wave-per-row kernels
  const int elemBlocks = (int)(((size_t)N * WDIM + 255) / 256);
  const int edgeWaveBlocks = (E + 7) / 8;

  auto gemm = [&](const __bf16* A, const __bf16* Bt, const float* bias, const float* res,
                  float* oF, __bf16* oBf, int M, int Kin, int Kout, int relu) {
    dim3 grid(M / 16, (Kout + 127) / 128);
    wmma_gemm_kernel<<<grid, 256, 0, stream>>>(A, Bt, bias, res, oF, oBf, M, Kin, Kout, relu);
  };

  // ---- 1) global token injection: Xcur = X + (g @ Wg + b)[batch]
  seg_sum_kernel<<<rowBlocks, 256, 0, stream>>>(X, bidx, gsum, gcnt, N);
  graph_linear_kernel<<<B, WDIM, 0, stream>>>(gsum, gcnt, g_token, Wg_w, Wg_b, g2);
  add_gather_kernel<<<elemBlocks, 256, 0, stream>>>(X, g2, bidx, Xcur, N);

  // ---- 2) PreNorm GIN-style MPNN
  layernorm_kernel<<<rowBlocks, 256, 0, stream>>>(Xcur, ln1_g, ln1_b, Xn, XnBf, N);
  gemm(XnBf, msgT, msg_b, nullptr, MfYf, nullptr, N, WDIM, WDIM, 0);            // msg
  scatter_agg_kernel<<<edgeWaveBlocks, 256, 0, stream>>>(MfYf, src, dst, agg, E);
  gin_pre_kernel<<<elemBlocks, 256, 0, stream>>>(Xn, agg, eps, XnBf, (size_t)N * WDIM);
  gemm(XnBf, linT, lin_b, Xcur, Xcur, nullptr, N, WDIM, WDIM, 1);               // X += relu(.)

  // ---- 3) PreNorm edge-restricted attention with segment softmax
  layernorm_kernel<<<rowBlocks, 256, 0, stream>>>(Xcur, ln2_g, ln2_b, nullptr, XnBf, N);
  gemm(XnBf, WqT, nullptr, nullptr, Qf,   nullptr, N, WDIM, WATTN, 0);
  gemm(XnBf, WkT, nullptr, nullptr, Kf,   nullptr, N, WDIM, WATTN, 0);
  gemm(XnBf, WyT, Wy_b,    nullptr, MfYf, nullptr, N, WDIM, WDIM,  0);          // Y
  edge_e_kernel<<<edgeWaveBlocks, 256, 0, stream>>>(Qf, Kf, Wphi, src, dst, eBf, scors, E);
  denom_kernel<<<(E + 255) / 256, 256, 0, stream>>>(scors, src, denom, E);
  attn_apply_kernel<<<E / 16, 256, 0, stream>>>(eBf, WeT, We_b, MfYf, scors, denom,
                                                src, dst, Xcur, E);

  // ---- 4) PreNorm FFN
  layernorm_kernel<<<rowBlocks, 256, 0, stream>>>(Xcur, ln3_g, ln3_b, nullptr, XnBf, N);
  gemm(XnBf, W1T, ffn_b1, nullptr, nullptr, HBf, N, WDIM, WHID, 1);             // H = relu(.)
  gemm(HBf,  W2T, ffn_b2, Xcur, Xcur, nullptr, N, WHID, WDIM, 0);               // X += H@W2+b2
}
